// MyAttention_layer_45715631898962
// MI455X (gfx1250) — matrix operand order repaired and checked by reading, not compiled, not run on hardware
//
#include <hip/hip_runtime.h>
#include <hip/hip_bf16.h>
#include <math.h>

typedef __attribute__((ext_vector_type(16))) _Float16 v16h;
typedef __attribute__((ext_vector_type(8)))  _Float16 v8h;
typedef __attribute__((ext_vector_type(8)))  float    v8f;
typedef __attribute__((ext_vector_type(4)))  float    f4;

constexpr int B_ = 2, INDIM = 256, POSDIM = 256, L_ = 2048, H_ = 8, DE_ = 64, DV_ = 64, RECEPT_ = 64;
constexpr int CDIM = INDIM + POSDIM; // 512

__device__ __forceinline__ v8f wmma16(v16h a, v16h b, v8f c) {
  // D = A(16x32 f16) * B(32x16 f16) + C(16x16 f32)
  return __builtin_amdgcn_wmma_f32_16x16x32_f16(false, a, false, b, (short)0, c, false, false);
}

__device__ __forceinline__ v16h pack16(v8h lo, v8h hi) {
  v16h r;
#pragma unroll
  for (int h = 0; h < 8; ++h) { r[h] = lo[h]; r[h + 8] = hi[h]; }
  return r;
}

// ---------------- prep: f32 -> f16 ----------------
__global__ void f32_to_f16_kernel(const float* __restrict__ src, _Float16* __restrict__ dst, int n) {
  int i = blockIdx.x * blockDim.x + threadIdx.x;
  int stride = gridDim.x * blockDim.x;
  for (; i < n; i += stride) dst[i] = (_Float16)src[i];
}

// ---------------- prep: xpT(B,L,512) = concat(x, pos) in f16, L-major ----------------
__global__ void build_xp_kernel(const float* __restrict__ x, _Float16* __restrict__ xpT) {
  int i = blockIdx.x * blockDim.x + threadIdx.x;
  int stride = gridDim.x * blockDim.x;
  const int n = B_ * L_ * CDIM;
  for (; i < n; i += stride) {
    int ch = i & (CDIM - 1);
    int l  = (i >> 9) & (L_ - 1);   // CDIM = 2^9
    int b  = i >> 20;               // 9 + 11
    float v;
    if (ch < INDIM) {
      v = x[((long long)b * INDIM + ch) * L_ + l];
    } else {
      int p = ch - INDIM;
      int isin = (p < POSDIM / 2);
      int idx = isin ? p : p - POSDIM / 2;
      float freq = (0.8f + powf(1.2f, (float)(idx + 1))) / (float)L_;
      float arg = 6.283185307179586f * (float)l * freq;
      v = isin ? sinf(arg) : cosf(arg);
    }
    xpT[i] = (_Float16)v;
  }
}

// ---------------- generic WMMA GEMM: Y = W(O,I) @ X, X given N-major as XT(N,ldx) ----------------
// All fragment traffic is contiguous b128 loads; transposed outputs give vectorized stores.
// mode bits: 1 = f16 store transposed YT(N,O), 2 = f32 store transposed, 4 = + f32 residual (transposed),
//            8 = exact GELU, 16 = f16 store channel-major Y(O,N) (for V)
__global__ void gemm_wmma_kernel(const _Float16* __restrict__ W,
                                 const _Float16* __restrict__ Xt, long long xStride, int ldx,
                                 _Float16* __restrict__ Yh, long long yhStride,
                                 float* __restrict__ Yf, long long yfStride,
                                 const float* __restrict__ Rf, long long rStride,
                                 int O, int I, int N, int mode) {
  const int lane = threadIdx.x & 31;
  const int wv   = threadIdx.x >> 5;
  const int nt   = N >> 4;
  const int tile = blockIdx.x * 4 + wv;
  const int o0 = (tile / nt) << 4;
  const int n0 = (tile % nt) << 4;
  const int g = lane >> 4, c = lane & 15;
  const _Float16* wrow = W + (long long)(o0 + c) * I;                                   // A row M = lane&15
  const _Float16* xrow = Xt + (long long)blockIdx.z * xStride + (long long)(n0 + c) * ldx; // B col per lane
  v8f acc = {};
  for (int k0 = 0; k0 < I; k0 += 32) {
    __builtin_prefetch(xrow + k0 + 64, 0, 0);   // -> global_prefetch_b8
    __builtin_prefetch(wrow + k0 + 64, 0, 0);
    // A (16x32): lane half g holds K = k0+16g..+15 contiguous  -> 2x b128
    v8h alo = *(const v8h*)(wrow + k0 + 16 * g);
    v8h ahi = *(const v8h*)(wrow + k0 + 16 * g + 8);
    // B (32x16): lane holds column n0+c, rows k0+16g..+15 contiguous in XT -> 2x b128
    v8h blo = *(const v8h*)(xrow + k0 + 16 * g);
    v8h bhi = *(const v8h*)(xrow + k0 + 16 * g + 8);
    acc = wmma16(pack16(alo, ahi), pack16(blo, bhi), acc);
  }
  if (mode & 16) { // channel-major f16 store (V): rows o0+r+8g, col n0+c
    _Float16* Y = Yh + (long long)blockIdx.z * yhStride;
#pragma unroll
    for (int r = 0; r < 8; ++r)
      Y[(long long)(o0 + r + 8 * g) * N + n0 + c] = (_Float16)acc[r];
    return; // wave-uniform
  }
  // transposed outputs: per-lane D rows o0+8g..+7 are contiguous in YT(N,O)
  const long long rowoff = (long long)(n0 + c) * O + o0 + 8 * g;
  v8f v = acc;
  if (mode & 4) {
    const float* R = Rf + (long long)blockIdx.z * rStride + rowoff;
    f4 r0 = *(const f4*)(R), r1 = *(const f4*)(R + 4);
#pragma unroll
    for (int r = 0; r < 4; ++r) { v[r] += r0[r]; v[r + 4] += r1[r]; }
  }
  if (mode & 8) {
#pragma unroll
    for (int r = 0; r < 8; ++r)
      v[r] = 0.5f * v[r] * (1.0f + erff(v[r] * 0.7071067811865476f));
  }
  if (mode & 2) {
    float* Y = Yf + (long long)blockIdx.z * yfStride + rowoff;
    f4 s0, s1;
#pragma unroll
    for (int r = 0; r < 4; ++r) { s0[r] = v[r]; s1[r] = v[r + 4]; }
    *(f4*)(Y) = s0; *(f4*)(Y + 4) = s1;
  }
  if (mode & 1) {
    _Float16* Y = Yh + (long long)blockIdx.z * yhStride + rowoff;
    v8h oh;
#pragma unroll
    for (int r = 0; r < 8; ++r) oh[r] = (_Float16)v[r];
    *(v8h*)(Y) = oh;
  }
}

// ---------------- Vsum: per (b,h,v) row sum over L (V is channel-major) ----------------
__global__ void vsum_kernel(const _Float16* __restrict__ vh, float* __restrict__ vsum) {
  const int lane = threadIdx.x & 31;
  const int row = blockIdx.x * 4 + (threadIdx.x >> 5); // B*H*DV = 1024 rows
  const _Float16* src = vh + (long long)row * L_;
  float s = 0.0f;
  for (int i = lane; i < L_; i += 32) s += (float)src[i];
#pragma unroll
  for (int off = 1; off < 32; off <<= 1) s += __shfl_xor(s, off);
  if (lane == 0) vsum[row] = s;
}

// ---------------- banded attention: all fragments via b128 loads ----------------
// out[l] = (sum_band (e^s - 1) v_m + Vsum) / (sum_band (e^s - 1) + L)
// qT,kT: (B, L, 512) L-major;  V: (B, 512, L) channel-major;  outT: (B, L, 512)
__global__ void attn_kernel(const _Float16* __restrict__ qT,
                            const _Float16* __restrict__ kT,
                            const _Float16* __restrict__ vh,
                            const float* __restrict__ vsum,
                            _Float16* __restrict__ outT) {
  __shared__ _Float16 Plds[4][16 * 32];   // per-wave P staging (D-layout -> A-layout)
  const int lane = threadIdx.x & 31;
  const int wv   = threadIdx.x >> 5;
  const int g = lane >> 4, c = lane & 15;
  const int l0 = (blockIdx.x * 4 + wv) << 4;
  const int head = blockIdx.y, b = blockIdx.z;
  const int bh = b * H_ + head;
  const _Float16* Kbase = kT + (long long)b * L_ * CDIM + head * DE_;
  const _Float16* V = vh + (long long)bh * DV_ * L_;

  // A = Q^T (16 l x 32 d): L-major layout makes the d-runs contiguous -> 4x b128
  v16h qa0, qa1;
  {
    const _Float16* Qrow = qT + ((long long)b * L_ + l0 + c) * CDIM + head * DE_;
    v8h t0 = *(const v8h*)(Qrow + 16 * g);
    v8h t1 = *(const v8h*)(Qrow + 16 * g + 8);
    v8h t2 = *(const v8h*)(Qrow + 32 + 16 * g);
    v8h t3 = *(const v8h*)(Qrow + 32 + 16 * g + 8);
    qa0 = pack16(t0, t1);
    qa1 = pack16(t2, t3);
  }

  v8f accv0 = {}, accv1 = {}, accv2 = {}, accv3 = {};
  float dsum[8];
#pragma unroll
  for (int r = 0; r < 8; ++r) dsum[r] = 0.0f;

  int mstart = l0 - RECEPT_; if (mstart < 0) mstart = 0; mstart &= ~31;
  int mend = l0 + 15 + RECEPT_ + 1; if (mend > L_) mend = L_;

  for (int m0 = mstart; m0 < mend; m0 += 32) {
#pragma unroll
    for (int sub = 0; sub < 2; ++sub) {
      const int mt = m0 + (sub << 4);
      // B = K (32 d x 16 m): lane's column mt+c, 16 contiguous d in kT -> 2x b128 per frag
      const _Float16* Km = Kbase + (long long)(mt + c) * CDIM;
      v8h k0lo = *(const v8h*)(Km + 16 * g);
      v8h k0hi = *(const v8h*)(Km + 16 * g + 8);
      v8h k1lo = *(const v8h*)(Km + 32 + 16 * g);
      v8h k1hi = *(const v8h*)(Km + 32 + 16 * g + 8);
      v8f s = {};
      s = wmma16(qa0, pack16(k0lo, k0hi), s);
      s = wmma16(qa1, pack16(k1lo, k1hi), s);
      const int m = mt + c;
#pragma unroll
      for (int r = 0; r < 8; ++r) {
        int l = l0 + r + 8 * g;
        int dlt = m - l; if (dlt < 0) dlt = -dlt;
        float p = (dlt <= RECEPT_) ? (expf(s[r] * 0.125f) - 1.0f) : 0.0f;
        dsum[r] += p;
        Plds[wv][(r + 8 * g) * 32 + (sub << 4) + c] = (_Float16)p;
      }
    }
    __builtin_amdgcn_wave_barrier();   // order LDS store -> load within wave
    // Reload P as A-fragment (16 l x 32 m): contiguous runs -> 2x ds b128
    v8h plo = *(const v8h*)(&Plds[wv][c * 32 + 16 * g]);
    v8h phi = *(const v8h*)(&Plds[wv][c * 32 + 16 * g + 8]);
    v16h pa = pack16(plo, phi);
    __builtin_amdgcn_wave_barrier();
    // B = V^T (32 m x 16 v): V channel-major -> 16 contiguous m per lane -> 2x b128 per tile
#pragma unroll
    for (int vt = 0; vt < 4; ++vt) {
      const _Float16* Vr = V + (long long)((vt << 4) + c) * L_ + m0 + 16 * g;
      v8h vlo = *(const v8h*)(Vr);
      v8h vhi = *(const v8h*)(Vr + 8);
      v16h vb = pack16(vlo, vhi);
      if (vt == 0) accv0 = wmma16(pa, vb, accv0);
      else if (vt == 1) accv1 = wmma16(pa, vb, accv1);
      else if (vt == 2) accv2 = wmma16(pa, vb, accv2);
      else accv3 = wmma16(pa, vb, accv3);
    }
  }

  // denom: row sums of P across the 16 columns (lanes of each half-wave)
#pragma unroll
  for (int r = 0; r < 8; ++r) {
#pragma unroll
    for (int off = 1; off < 16; off <<= 1)
      dsum[r] += __shfl_xor(dsum[r], off);
  }

  const float* VS = vsum + bh * DV_;
#pragma unroll
  for (int vt = 0; vt < 4; ++vt) {
    v8f a = (vt == 0) ? accv0 : (vt == 1) ? accv1 : (vt == 2) ? accv2 : accv3;
    int v = (vt << 4) + c;
    float vs = VS[v];
#pragma unroll
    for (int r = 0; r < 8; ++r) {
      float o = (a[r] + vs) / (dsum[r] + (float)L_);
      outT[((long long)b * L_ + l0 + r + 8 * g) * CDIM + head * DE_ + v] = (_Float16)o;
    }
  }
}

// ---------------- LayerNorm over 64 channels, wave per (b,l); L-major in/out ----------------
__global__ void ln_kernel(const float* __restrict__ srcT,      // (B*L, 64)
                          const float* __restrict__ gamma,
                          const float* __restrict__ beta,
                          float* __restrict__ outT,            // (B*L, 64) f32 (if out_cm==null)
                          _Float16* __restrict__ outhT,        // (B*L, 64) f16 (if out_cm==null)
                          float* __restrict__ out_cm) {        // (B, 64, L) final output
  const int lane = threadIdx.x & 31;
  const int idx = blockIdx.x * 4 + (threadIdx.x >> 5); // over B*L
  const float* base = srcT + (long long)idx * DV_;
  float v0 = base[lane];
  float v1 = base[lane + 32];
  float s = v0 + v1;
#pragma unroll
  for (int off = 1; off < 32; off <<= 1) s += __shfl_xor(s, off);
  float mu = s * (1.0f / DV_);
  float d0 = v0 - mu, d1 = v1 - mu;
  float q = d0 * d0 + d1 * d1;
#pragma unroll
  for (int off = 1; off < 32; off <<= 1) q += __shfl_xor(q, off);
  float inv = rsqrtf(q * (1.0f / DV_) + 1e-5f);
  float y0 = d0 * inv * gamma[lane] + beta[lane];
  float y1 = d1 * inv * gamma[lane + 32] + beta[lane + 32];
  if (out_cm) {
    const int b = idx >> 11, l = idx & (L_ - 1);
    out_cm[((long long)b * DV_ + lane) * L_ + l] = y0;
    out_cm[((long long)b * DV_ + lane + 32) * L_ + l] = y1;
  } else {
    outT[(long long)idx * DV_ + lane] = y0;
    outT[(long long)idx * DV_ + lane + 32] = y1;
    outhT[(long long)idx * DV_ + lane] = (_Float16)y0;
    outhT[(long long)idx * DV_ + lane + 32] = (_Float16)y1;
  }
}

extern "C" void kernel_launch(void* const* d_in, const int* in_sizes, int n_in,
                              void* d_out, int out_size, void* d_ws, size_t ws_size,
                              hipStream_t stream) {
  (void)in_sizes; (void)n_in; (void)out_size; (void)ws_size;
  const float* x    = (const float*)d_in[0];
  const float* Wq   = (const float*)d_in[1];
  const float* Wk   = (const float*)d_in[2];
  const float* Wv   = (const float*)d_in[3];
  const float* Wres = (const float*)d_in[4];
  const float* Wc   = (const float*)d_in[5];
  const float* ln1g = (const float*)d_in[6];
  const float* ln1b = (const float*)d_in[7];
  const float* W1   = (const float*)d_in[8];
  const float* W2   = (const float*)d_in[9];
  const float* ln2g = (const float*)d_in[10];
  const float* ln2b = (const float*)d_in[11];
  float* out = (float*)d_out;

  char* ws = (char*)d_ws;
  size_t off = 0;
  auto alloc = [&](size_t bytes) -> void* {
    off = (off + 255) & ~(size_t)255;
    void* p = ws + off;
    off += bytes;
    return p;
  };
  const long long BL = (long long)B_ * L_;
  _Float16* xpT    = (_Float16*)alloc((size_t)BL * CDIM * 2);   // (B,L,512)
  _Float16* Wq_h   = (_Float16*)alloc((size_t)CDIM * CDIM * 2);
  _Float16* Wk_h   = (_Float16*)alloc((size_t)CDIM * CDIM * 2);
  _Float16* Wv_h   = (_Float16*)alloc((size_t)CDIM * CDIM * 2);
  _Float16* Wres_h = (_Float16*)alloc((size_t)DV_ * INDIM * 2);
  _Float16* Wc_h   = (_Float16*)alloc((size_t)DV_ * (H_ * DV_) * 2);
  _Float16* W1_h   = (_Float16*)alloc((size_t)4 * DV_ * DV_ * 2);
  _Float16* W2_h   = (_Float16*)alloc((size_t)DV_ * 4 * DV_ * 2);
  _Float16* qTb    = (_Float16*)alloc((size_t)BL * CDIM * 2);   // (B,L,512)
  _Float16* kTb    = (_Float16*)alloc((size_t)BL * CDIM * 2);   // (B,L,512)
  _Float16* vh     = (_Float16*)alloc((size_t)BL * CDIM * 2);   // (B,512,L)
  float*    vsum   = (float*)   alloc((size_t)B_ * H_ * DV_ * 4);
  _Float16* attnT  = (_Float16*)alloc((size_t)BL * CDIM * 2);   // (B,L,512)
  float*    residT = (float*)   alloc((size_t)BL * DV_ * 4);    // (B,L,64)
  float*    predT  = (float*)   alloc((size_t)BL * DV_ * 4);
  float*    predlnT  = (float*) alloc((size_t)BL * DV_ * 4);
  _Float16* predlnhT = (_Float16*)alloc((size_t)BL * DV_ * 2);
  _Float16* ffhT   = (_Float16*)alloc((size_t)BL * 4 * DV_ * 2); // (B,L,256)
  float*    pred2T = (float*)   alloc((size_t)BL * DV_ * 4);

  // --- prep ---
  f32_to_f16_kernel<<<256, 256, 0, stream>>>(Wq,   Wq_h,   CDIM * CDIM);
  f32_to_f16_kernel<<<256, 256, 0, stream>>>(Wk,   Wk_h,   CDIM * CDIM);
  f32_to_f16_kernel<<<256, 256, 0, stream>>>(Wv,   Wv_h,   CDIM * CDIM);
  f32_to_f16_kernel<<<64,  256, 0, stream>>>(Wres, Wres_h, DV_ * INDIM);
  f32_to_f16_kernel<<<64,  256, 0, stream>>>(Wc,   Wc_h,   DV_ * H_ * DV_);
  f32_to_f16_kernel<<<64,  256, 0, stream>>>(W1,   W1_h,   4 * DV_ * DV_);
  f32_to_f16_kernel<<<64,  256, 0, stream>>>(W2,   W2_h,   DV_ * 4 * DV_);
  build_xp_kernel<<<1024, 256, 0, stream>>>(x, xpT);

  const long long xpS = (long long)L_ * CDIM;   // batch stride of (L,512) tensors
  const long long dS  = (long long)L_ * DV_;    // batch stride of (L,64) tensors
  const long long fS  = (long long)L_ * 4 * DV_;

  dim3 blk(128);
  // --- Q,K: transposed f16 out (L,512); V: channel-major f16 out (512,L) ---
  gemm_wmma_kernel<<<dim3(1024, 1, B_), blk, 0, stream>>>(Wq_h, xpT, xpS, CDIM, qTb, xpS, nullptr, 0, nullptr, 0, CDIM, CDIM, L_, 1);
  gemm_wmma_kernel<<<dim3(1024, 1, B_), blk, 0, stream>>>(Wk_h, xpT, xpS, CDIM, kTb, xpS, nullptr, 0, nullptr, 0, CDIM, CDIM, L_, 1);
  gemm_wmma_kernel<<<dim3(1024, 1, B_), blk, 0, stream>>>(Wv_h, xpT, xpS, CDIM, vh,  xpS, nullptr, 0, nullptr, 0, CDIM, CDIM, L_, 16);
  // --- residual: 64x256 @ x part of xpT (ldx=512, I=256), f32 transposed out ---
  gemm_wmma_kernel<<<dim3(128, 1, B_), blk, 0, stream>>>(Wres_h, xpT, xpS, CDIM, nullptr, 0, residT, dS, nullptr, 0, DV_, INDIM, L_, 2);

  // --- Vsum + banded attention ---
  vsum_kernel<<<256, 128, 0, stream>>>(vh, vsum);
  attn_kernel<<<dim3(L_ / 64, H_, B_), blk, 0, stream>>>(qTb, kTb, vh, vsum, attnT);

  // --- Wc + residual -> predT (f32, (L,64)) ---
  gemm_wmma_kernel<<<dim3(128, 1, B_), blk, 0, stream>>>(Wc_h, attnT, xpS, CDIM, nullptr, 0, predT, dS, residT, dS, DV_, H_ * DV_, L_, 2 | 4);
  // --- LN1 -> predlnT + predlnhT ---
  ln_kernel<<<1024, 128, 0, stream>>>(predT, ln1g, ln1b, predlnT, predlnhT, nullptr);
  // --- FFN ---
  gemm_wmma_kernel<<<dim3(512, 1, B_), blk, 0, stream>>>(W1_h, predlnhT, dS, DV_, ffhT, fS, nullptr, 0, nullptr, 0, 4 * DV_, DV_, L_, 1 | 8);
  gemm_wmma_kernel<<<dim3(128, 1, B_), blk, 0, stream>>>(W2_h, ffhT, fS, 4 * DV_, nullptr, 0, pred2T, dS, predlnT, dS, DV_, 4 * DV_, L_, 2 | 4);
  // --- LN2 -> final (B,64,L) output ---
  ln_kernel<<<1024, 128, 0, stream>>>(pred2T, ln2g, ln2b, nullptr, nullptr, out);
}
